// SFNN_5111011083122
// MI455X (gfx1250) — matrix operand order repaired and checked by reading, not compiled, not run on hardware
//
#include <hip/hip_runtime.h>

// SFNN graph-network step for MI455X (gfx1250, wave32, WMMA).
// Dominant work (edge GRU: 2x [768x32]@[32x96] GEMMs per source node, 7.2 GFLOP,
// + full read/modify/write of e [75.5 MB]) is fused into one kernel whose inner
// products run on v_wmma_f32_16x16x32_f16. Gate nonlinearities use raw
// transcendental-unit ops (v_exp/v_rcp/v_tanh) instead of IEEE-division sigmoids.

#define NN   768
#define DE   32
#define NT   8
#define DIN  64
#define DACT 16
#define G3   96      // 3*DE
#define EIN  65      // 2*DE+1

typedef __attribute__((ext_vector_type(16))) _Float16 v16h;
typedef __attribute__((ext_vector_type(8)))  _Float16 v8h;
typedef __attribute__((ext_vector_type(8)))  float    v8f;
typedef __attribute__((ext_vector_type(4)))  float    v4f;

// Fast transcendental helpers: v_exp_f32 + v_rcp_f32 (no div fixup sequence),
// v_tanh_f32 when the toolchain exposes it.
__device__ __forceinline__ float fast_rcp_(float x) {
#if __has_builtin(__builtin_amdgcn_rcpf)
  return __builtin_amdgcn_rcpf(x);
#else
  return 1.0f / x;
#endif
}
__device__ __forceinline__ float sigmoidf_(float x) {
  return fast_rcp_(1.0f + __expf(-x));
}
__device__ __forceinline__ float fast_tanh_(float x) {
#if __has_builtin(__builtin_amdgcn_tanhf)
  return __builtin_amdgcn_tanhf(x);
#else
  // tanh(x) = 1 - 2/(exp(2x)+1)
  return 1.0f - 2.0f * fast_rcp_(__expf(2.0f * x) + 1.0f);
#endif
}

// ---------------------------------------------------------------------------
// x[j,d] = sum_i y[i,d] * A[i,j] * e[i,j,d]  (+ obs / one-hot action overrides)
// One block per j; 8 i-groups x 32 d-lanes; e rows (128B) read coalesced.
// ---------------------------------------------------------------------------
__global__ void msg_kernel(const float* __restrict__ y, const float* __restrict__ e,
                           const float* __restrict__ A, const float* __restrict__ obs,
                           const int* __restrict__ a_prev, int apply_act,
                           float* __restrict__ x) {
  __shared__ float red[8][DE];
  const int j = blockIdx.x;
  const int t = threadIdx.x;
  const int d = t & 31, ig = t >> 5;
  float acc = 0.f;
  for (int i = ig; i < NN; i += 8) {
    float a = A[i * NN + j];  // wave-uniform broadcast
    acc += y[i * DE + d] * a * e[((size_t)i * NN + j) * DE + d];
  }
  red[ig][d] = acc;
  __syncthreads();
  if (t < DE) {
    float s = 0.f;
#pragma unroll
    for (int k = 0; k < 8; ++k) s += red[k][t];
    float v = (j < DIN) ? obs[j] : s;
    if (apply_act && j >= NN - DACT && t == 0)
      v = (j - (NN - DACT) == a_prev[0]) ? 1.0f : 0.0f;
    x[j * DE + t] = v;
  }
}

// ---------------------------------------------------------------------------
// Node GRU cell with h=0: y[n,d] = (1-sig(iz)) * tanh(inn + sig(ir)*bn).
// One wave per node (8 nodes / 256-thread block); thread d owns 3 gate rows.
// ---------------------------------------------------------------------------
__global__ void node_kernel(const float* __restrict__ x, const int* __restrict__ types,
                            const float* __restrict__ W, const float* __restrict__ b,
                            const float* __restrict__ bn,
                            float* __restrict__ yout, _Float16* __restrict__ y16) {
  const int t = threadIdx.x;
  const int node = blockIdx.x * 8 + (t >> 5);
  const int d = t & 31;
  const int ty = types[node];
  const float* xr = x + node * DE;
  const float* Wb = W + (size_t)ty * G3 * DE;
  float ir = b[ty * G3 + d], iz = b[ty * G3 + DE + d], inn = b[ty * G3 + 2 * DE + d];
#pragma unroll 8
  for (int k = 0; k < DE; ++k) {
    float xv = xr[k];
    ir  += Wb[(size_t)d * DE + k] * xv;
    iz  += Wb[(size_t)(DE + d) * DE + k] * xv;
    inn += Wb[(size_t)(2 * DE + d) * DE + k] * xv;
  }
  float rg = sigmoidf_(ir), zg = sigmoidf_(iz);
  float nn = fast_tanh_(inn + rg * bn[ty * DE + d]);
  float yv = (1.0f - zg) * nn;
  yout[node * DE + d] = yv;
  if (y16) y16[node * DE + d] = (_Float16)yv;
}

// ---------------------------------------------------------------------------
// Per-type f16 copies of edge weights, laid out [type][gate g][k] contiguous-k
// so a B-fragment lane (g = tile*16 + lane&15) reads two 8-half runs, k=8*hi..+7 and 16+8*hi..+7.
// ---------------------------------------------------------------------------
__global__ void cvtw_kernel(const float* __restrict__ Wih, const float* __restrict__ Whh,
                            _Float16* __restrict__ WihT16, _Float16* __restrict__ WhhT16) {
  int idx = blockIdx.x * blockDim.x + threadIdx.x;  // over NT*G3*DE
  if (idx >= NT * G3 * DE) return;
  int k = idx & 31;
  int g = (idx >> 5) % G3;
  int ty = idx / (G3 * DE);
  WihT16[idx] = (_Float16)Wih[((size_t)ty * G3 + g) * EIN + DE + k];  // y_post slice
  WhhT16[idx] = (_Float16)Whh[idx];
}

__global__ void argmax_kernel(const float* __restrict__ ypost, float* __restrict__ out) {
  if (threadIdx.x == 0) {
    int best = 0;
    float bv = ypost[(NN - DACT) * DE];
    for (int k = 1; k < DACT; ++k) {
      float v = ypost[(NN - DACT + k) * DE];
      if (v > bv) { bv = v; best = k; }  // strict > keeps first max (jnp.argmax)
    }
    out[0] = (float)best;
  }
}

// ---------------------------------------------------------------------------
// Fused edge GRU. One block per source node i (8 waves). Per 16-row j-tile:
//   IG tile = Y_post[j0:j0+16, :] @ Wih_slice^T   (WMMA f16, K=32 in one op)
//   HG tile = e[i, j0:j0+16, :]   @ Whh^T          (WMMA f16)
// then per-lane gate math + residual update of e, single pass over e.
// Fragment layouts follow CDNA5 ISA §7.12.2 (16-bit A 16x32,
// B 32x16 by K-striping, f32 C/D 16x16).
// ---------------------------------------------------------------------------
__global__ __launch_bounds__(256)
void edge_kernel(const float* __restrict__ e, const int* __restrict__ types,
                 const float* __restrict__ y_prev, const _Float16* __restrict__ y16,
                 const float* __restrict__ Wih, const float* __restrict__ edge_b,
                 const float* __restrict__ edge_bn, const float* __restrict__ rscal,
                 const float* __restrict__ lr,
                 const _Float16* __restrict__ WihT16, const _Float16* __restrict__ WhhT16,
                 float* __restrict__ eout) {
  __shared__ float base_s[G3];
  __shared__ float bn_s[DE];
  const int i = blockIdx.x;
  const int t = threadIdx.x;
  const int ty = types[i];

  // base_i[g] = dot(Wih[ty,g,0:DE], y_prev[i]) + Wih[ty,g,2DE]*r + edge_b[ty,g]
  if (t < G3) {
    const float* wr = Wih + ((size_t)ty * G3 + t) * EIN;
    const float* yp = y_prev + i * DE;
    float acc = edge_b[ty * G3 + t] + wr[2 * DE] * rscal[0];
#pragma unroll 8
    for (int k = 0; k < DE; ++k) acc += wr[k] * yp[k];
    base_s[t] = acc;
  }
  if (t < DE) bn_s[t] = edge_bn[ty * DE + t];
  __syncthreads();

  const int wave = t >> 5;
  const int lane = t & 31;
  const int m = lane & 15;   // A row / C column within tile
  const int hi = lane >> 4;  // lane half-group
  const float scale = 0.01f * lr[0];

  // Preload the 12 B-fragments (6 gate tiles x {Wih, Whh}) for this node type.
  const v8h* WihV = (const v8h*)(WihT16 + (size_t)ty * G3 * DE);
  const v8h* WhhV = (const v8h*)(WhhT16 + (size_t)ty * G3 * DE);
  v16h Bih[6], Bhh[6];
#pragma unroll
  for (int gt = 0; gt < 6; ++gt) {
    const int g = gt * 16 + m;                     // B column = gate index
    v8h a0 = WihV[g * 4 + hi], a1 = WihV[g * 4 + 2 + hi];  // K=8*hi..+7, 16+8*hi..+7
    v8h b0 = WhhV[g * 4 + hi], b1 = WhhV[g * 4 + 2 + hi];
#pragma unroll
    for (int s = 0; s < 8; ++s) {
      Bih[gt][s] = a0[s]; Bih[gt][8 + s] = a1[s];
      Bhh[gt][s] = b0[s]; Bhh[gt][8 + s] = b1[s];
    }
  }

  for (int jt = wave; jt < NN / 16; jt += 8) {   // wave-uniform: EXEC all-1s for WMMA
    const int j0 = jt * 16;

    // A-fragment from y_post (f16): slots 0..7 -> K=8*hi+s, 8..15 -> K=16+8*hi+s
    const v8h* yr = (const v8h*)(y16 + (j0 + m) * DE);
    v16h Ay;
    {
      v8h c0 = yr[hi], c1 = yr[2 + hi];
#pragma unroll
      for (int s = 0; s < 8; ++s) { Ay[s] = c0[s]; Ay[8 + s] = c1[s]; }
    }
    // A-fragment from e[i] (f32 -> f16)
    const v4f* er4 = (const v4f*)(e + ((size_t)i * NN + j0 + m) * DE);
    v16h Ae;
    {
      v4f c0 = er4[2 * hi], c1 = er4[2 * hi + 1];
      v4f c2 = er4[4 + 2 * hi], c3 = er4[4 + 2 * hi + 1];
#pragma unroll
      for (int s = 0; s < 4; ++s) {
        Ae[s]      = (_Float16)c0[s];
        Ae[4 + s]  = (_Float16)c1[s];
        Ae[8 + s]  = (_Float16)c2[s];
        Ae[12 + s] = (_Float16)c3[s];
      }
    }

    v8f IG[6], HG[6];
#pragma unroll
    for (int gt = 0; gt < 6; ++gt) {
      v8f z = {};
      IG[gt] = __builtin_amdgcn_wmma_f32_16x16x32_f16(false, Ay, false, Bih[gt],
                                                      (short)0, z, false, false);
      HG[gt] = __builtin_amdgcn_wmma_f32_16x16x32_f16(false, Ae, false, Bhh[gt],
                                                      (short)0, z, false, false);
    }

    // Gate math + residual update. Lane owns (j = j0+8*hi+v, d = h*16+m).
#pragma unroll
    for (int v = 0; v < 8; ++v) {
      const int j = j0 + 8 * hi + v;
      const float* er = e + ((size_t)i * NN + j) * DE;
      float* eo = eout + ((size_t)i * NN + j) * DE;
#pragma unroll
      for (int h = 0; h < 2; ++h) {
        const int d = h * 16 + m;
        float rg = sigmoidf_(IG[h][v] + base_s[d] + HG[h][v]);
        float zg = sigmoidf_(IG[2 + h][v] + base_s[DE + d] + HG[2 + h][v]);
        float nn = fast_tanh_(IG[4 + h][v] + base_s[2 * DE + d] +
                              rg * (HG[4 + h][v] + bn_s[d]));
        float ef = er[d];
        float outv = nn + zg * (ef - nn);
        eo[d] = ef + scale * outv;
      }
    }
  }
}

// ---------------------------------------------------------------------------
extern "C" void kernel_launch(void* const* d_in, const int* in_sizes, int n_in,
                              void* d_out, int out_size, void* d_ws, size_t ws_size,
                              hipStream_t stream) {
  (void)in_sizes; (void)n_in; (void)out_size; (void)ws_size;
  const float* obs      = (const float*)d_in[0];
  const int*   types    = (const int*)d_in[1];
  const float* e        = (const float*)d_in[2];
  const float* A        = (const float*)d_in[3];
  const float* r        = (const float*)d_in[4];
  const float* y_prev   = (const float*)d_in[5];
  const int*   a_prev   = (const int*)d_in[6];
  const float* node_W   = (const float*)d_in[7];
  const float* node_b   = (const float*)d_in[8];
  const float* node_bn  = (const float*)d_in[9];
  const float* edge_Wih = (const float*)d_in[10];
  const float* edge_Whh = (const float*)d_in[11];
  const float* edge_b   = (const float*)d_in[12];
  const float* edge_bn  = (const float*)d_in[13];
  const float* lr       = (const float*)d_in[14];

  float* out   = (float*)d_out;
  float* e_new = out + 1;                                // [N,N,DE]
  float* y_out = out + 1 + (size_t)NN * NN * DE;         // [N,DE]

  // Workspace carve-out (256B-aligned slabs, ~440 KB total).
  char* ws = (char*)d_ws;
  size_t off = 0;
  auto alloc = [&](size_t bytes) -> void* {
    off = (off + 255) & ~(size_t)255;
    void* p = ws + off;
    off += bytes;
    return p;
  };
  float*    x       = (float*)alloc((size_t)NN * DE * 4);
  float*    yA      = (float*)alloc((size_t)NN * DE * 4);
  float*    yB      = (float*)alloc((size_t)NN * DE * 4);
  _Float16* y16     = (_Float16*)alloc((size_t)NN * DE * 2);
  _Float16* WihT16  = (_Float16*)alloc((size_t)NT * G3 * DE * 2);
  _Float16* WhhT16  = (_Float16*)alloc((size_t)NT * G3 * DE * 2);

  // f16 weight staging for the WMMA edge kernel (independent of passes).
  cvtw_kernel<<<(NT * G3 * DE + 255) / 256, 256, 0, stream>>>(edge_Wih, edge_Whh,
                                                              WihT16, WhhT16);

  // Pass 0: obs override only.
  msg_kernel<<<NN, 256, 0, stream>>>(y_prev, e, A, obs, a_prev, 0, x);
  node_kernel<<<NN / 8, 256, 0, stream>>>(x, types, node_W, node_b, node_bn,
                                          yA, (_Float16*)nullptr);
  // Pass 1: obs + one-hot action override.
  msg_kernel<<<NN, 256, 0, stream>>>(yA, e, A, obs, a_prev, 1, x);
  node_kernel<<<NN / 8, 256, 0, stream>>>(x, types, node_W, node_b, node_bn,
                                          yB, (_Float16*)nullptr);
  // Pass 2 (final): y_post straight into d_out tail + f16 copy for WMMA.
  msg_kernel<<<NN, 256, 0, stream>>>(yB, e, A, obs, a_prev, 1, x);
  node_kernel<<<NN / 8, 256, 0, stream>>>(x, types, node_W, node_b, node_bn,
                                          y_out, y16);

  argmax_kernel<<<1, 32, 0, stream>>>(y_out, out);

  // The big one: fused edge GRU on the matrix units, single pass over e.
  edge_kernel<<<NN, 256, 0, stream>>>(e, types, y_prev, y16, edge_Wih, edge_b,
                                      edge_bn, r, lr, WihT16, WhhT16, e_new);
}
